// Attention_73581379715274
// MI455X (gfx1250) — matrix operand order repaired and checked by reading, not compiled, not run on hardware
//
#include <hip/hip_runtime.h>
#include <hip/hip_bf16.h>

// ---------------------------------------------------------------------------
// Attention block for MI455X (gfx1250): bf16 WMMA everywhere, f32 softmax.
//   B=1, S=2048, D=2048, H=32, KVH=8, HD=64
// ---------------------------------------------------------------------------

#define S_LEN   2048
#define D_MODEL 2048
#define NHEADS  32
#define KVHEADS 8
#define HEADDIM 64
#define KVDIM   (KVHEADS * HEADDIM)   // 512

typedef __bf16 bf16_t;
typedef __attribute__((ext_vector_type(16))) __bf16 v16bf;
typedef __attribute__((ext_vector_type(8)))  __bf16 v8bf;
typedef __attribute__((ext_vector_type(8)))  float  v8f;
typedef __attribute__((ext_vector_type(4)))  int      v4i;
typedef __attribute__((ext_vector_type(8)))  int      v8i;
typedef __attribute__((ext_vector_type(4)))  unsigned v4u;

__device__ __forceinline__ bf16_t f2bf(float f) {
  unsigned u = __builtin_bit_cast(unsigned, f);
  unsigned r = u + 0x7FFFu + ((u >> 16) & 1u);   // round-to-nearest-even
  unsigned short us = (unsigned short)(r >> 16);
  return __builtin_bit_cast(bf16_t, us);
}
__device__ __forceinline__ float bf2f(bf16_t b) {
  unsigned short us = __builtin_bit_cast(unsigned short, b);
  unsigned u = ((unsigned)us) << 16;
  return __builtin_bit_cast(float, u);
}

__device__ __forceinline__ v8f wmma_bf16(v16bf a, v16bf b, v8f c) {
  return __builtin_amdgcn_wmma_f32_16x16x32_bf16(
      /*neg_a=*/false, a, /*neg_b=*/false, b,
      /*c_mod=*/(short)0, c, /*reuse_a=*/false, /*reuse_b=*/false);
}

// A fragment: 16x32 bf16 tile, source row-major (row stride ld halves).
// Lanes 0-15: M=lane,  K={0..7,16..23}; lanes 16-31: M=lane-16, K={8..15,24..31}.
__device__ __forceinline__ v16bf load_a16x32(const bf16_t* base, int ld) {
  const int lane = threadIdx.x & 31;
  const int m    = lane & 15;
  const int kb   = (lane & 16) ? 8 : 0;
  const bf16_t* row = base + (size_t)m * ld + kb;
  v16bf a;
  __builtin_memcpy(&a, __builtin_assume_aligned(row, 16), 16);
  __builtin_memcpy((char*)&a + 16, __builtin_assume_aligned(row + 16, 16), 16);
  return a;
}

// B fragment: K=32 x N=16 where the source is stored N-major:
//   B[k][n] = base[n*ld + k]   (each lane's K-values are two contiguous runs of 8)
// Lanes 0-15: N=lane, K={0..7,16..23}; lanes 16-31: N=lane-16, K={8..15,24..31}.
__device__ __forceinline__ v16bf load_b_nmajor(const bf16_t* base, int ld) {
  const int lane = threadIdx.x & 31;
  const int nn   = lane & 15;
  const int kb   = (lane & 16) ? 8 : 0;
  const bf16_t* row = base + (size_t)nn * ld + kb;
  v16bf b;
  __builtin_memcpy(&b, __builtin_assume_aligned(row, 16), 16);
  __builtin_memcpy((char*)&b + 16, __builtin_assume_aligned(row + 16, 16), 16);
  return b;
}

// ---- direct global->LDS async staging (CDNA5 ASYNCcnt path) ---------------
#if __has_builtin(__builtin_amdgcn_global_load_async_to_lds_b128)
#define HAVE_ASYNC_LDS 1
typedef __attribute__((address_space(1))) v4i* as1_v4i;
typedef __attribute__((address_space(3))) v4i* as3_v4i;
__device__ __forceinline__ void async_cp16(const bf16_t* g, bf16_t* l) {
  __builtin_amdgcn_global_load_async_to_lds_b128(
      (as1_v4i)(bf16_t*)g, (as3_v4i)l, /*offset=*/0, /*cpol=*/0);
}
#else
#define HAVE_ASYNC_LDS 0
#endif

// ---- Tensor Data Mover staging (CDNA5 TENSORcnt path) ---------------------
#if __has_builtin(__builtin_amdgcn_tensor_load_to_lds)
#define HAVE_TDM 1
// 2-D tile load: rows x 64 bf16 elements from a row-major [*, K] matrix into
// contiguous LDS.  D# per CDNA5 ISA §8.3-8.6 (group0: ctrl/lds/global addr,
// group1: dims/strides, groups 2-3 unused for 2-D).
// This toolchain's builtin takes 6 args (probe: "expected 6"):
//   (v4u g0, v8i g1, v4i g2, v4i g3, v8i extra, i32 cpol)
__device__ __forceinline__ void tdm_load_tile(bf16_t* lds, const bf16_t* g,
                                              int K_elems, int rows) {
  unsigned laddr = (unsigned)(unsigned long long)
      (__attribute__((address_space(3))) char*)(void*)lds;
  unsigned long long ga = (unsigned long long)g;
  v4u g0;
  g0[0] = 1u;                                           // count=1 (valid D#)
  g0[1] = laddr;                                        // lds_addr [63:32]
  g0[2] = (unsigned)(ga & 0xFFFFFFFFu);                 // global_addr lo
  g0[3] = (unsigned)((ga >> 32) & 0x01FFFFFFu)          // global_addr [56:32]
        | (2u << 30);                                   // type=2 ("image")
  const unsigned td0 = (unsigned)K_elems;               // tensor_dim0
  const unsigned td1 = (unsigned)rows;                  // tensor_dim1
  v8i g1;
  g1[0] = (int)(1u << 16);                              // data_size=1 -> 2 bytes
  g1[1] = (int)((td0 & 0xFFFFu) << 16);                 // tensor_dim0[15:0]
  g1[2] = (int)((td0 >> 16) | ((td1 & 0xFFFFu) << 16)); // td0 hi | td1 lo
  g1[3] = (int)((td1 >> 16) | (64u << 16));             // td1 hi | tile_dim0=64
  g1[4] = (int)((unsigned)rows);                        // tile_dim1=rows, tile_dim2=0
  g1[5] = (int)td0;                                     // tensor_dim0_stride = K
  g1[6] = 0;
  g1[7] = 0;
  v4i z4 = {0, 0, 0, 0};
  v8i z8 = {0, 0, 0, 0, 0, 0, 0, 0};
  __builtin_amdgcn_tensor_load_to_lds(g0, g1, z4, z4, z8, /*cpol=*/0);
}
__device__ __forceinline__ void tdm_wait0() {
#if __has_builtin(__builtin_amdgcn_s_wait_tensorcnt)
  __builtin_amdgcn_s_wait_tensorcnt(0);
#else
  asm volatile("s_wait_tensorcnt 0" ::: "memory");
#endif
}
#else
#define HAVE_TDM 0
#endif

// ---------------------------------------------------------------------------
// Elementwise / packing kernels
// ---------------------------------------------------------------------------
__global__ void f32_to_bf16_kernel(const float* __restrict__ src,
                                   bf16_t* __restrict__ dst, int n) {
  int i = blockIdx.x * blockDim.x + threadIdx.x;
  if (i < n) dst[i] = f2bf(src[i]);
}

// src f32 [R][C]  ->  dst bf16 [C][R]  (LDS-tiled, both sides coalesced)
__global__ __launch_bounds__(256) void transpose_f32_to_bf16_kernel(
    const float* __restrict__ src, bf16_t* __restrict__ dst, int R, int C) {
  __shared__ bf16_t tile[32][33];
  const int bx = blockIdx.x * 32;       // C base
  const int by = blockIdx.y * 32;       // R base
  const int tx = threadIdx.x & 31;
  const int ty = threadIdx.x >> 5;      // 0..7
#pragma unroll
  for (int i = 0; i < 32; i += 8)
    tile[ty + i][tx] = f2bf(src[(size_t)(by + ty + i) * C + bx + tx]);
  __syncthreads();
#pragma unroll
  for (int i = 0; i < 32; i += 8)
    dst[(size_t)(bx + ty + i) * R + by + tx] = tile[tx][ty + i];
}

// src: bf16 [S][nh*64]; dst: bf16 [nh][S][64] with RoPE applied.
__global__ void rope_pack_kernel(const bf16_t* __restrict__ src,
                                 const float* __restrict__ cs,
                                 const float* __restrict__ sn,
                                 bf16_t* __restrict__ dst, int nh, int total) {
  int idx = blockIdx.x * blockDim.x + threadIdx.x;
  if (idx >= total) return;
  int i   = idx & 31;          // rotary pair index 0..31
  int rem = idx >> 5;
  int hh  = rem % nh;
  int s   = rem / nh;
  float c  = cs[s * 32 + i];
  float si = sn[s * 32 + i];
  size_t so = (size_t)s * (nh * HEADDIM) + hh * HEADDIM + 2 * i;
  float t0 = bf2f(src[so]);
  float t1 = bf2f(src[so + 1]);
  size_t o = ((size_t)hh * S_LEN + s) * HEADDIM + 2 * i;
  dst[o]     = f2bf(t0 * c - t1 * si);
  dst[o + 1] = f2bf(t0 * si + t1 * c);
}

// src: bf16 [S][512]; dst: bf16 [KVH][64][S] (d-major so P*V B-frags are contiguous)
__global__ void pack_vT_kernel(const bf16_t* __restrict__ src,
                               bf16_t* __restrict__ dst, int total) {
  int idx = blockIdx.x * blockDim.x + threadIdx.x;
  if (idx >= total) return;
  int s   = idx & (S_LEN - 1);
  int rem = idx >> 11;
  int d   = rem & (HEADDIM - 1);
  int kvh = rem >> 6;
  dst[((size_t)(kvh * HEADDIM + d)) * S_LEN + s] =
      src[(size_t)s * KVDIM + kvh * HEADDIM + d];
}

// fw f32 [64][64] -> fwT bf16 [64][64], fwT[n][k] = fw[k][n]
__global__ void pack_fwT_kernel(const float* __restrict__ fw,
                                bf16_t* __restrict__ fwT) {
  int idx = blockIdx.x * blockDim.x + threadIdx.x;
  if (idx >= HEADDIM * HEADDIM) return;
  int k = idx >> 6, n = idx & 63;
  fwT[(size_t)n * HEADDIM + k] = f2bf(fw[(size_t)k * HEADDIM + n]);
}

// ---------------------------------------------------------------------------
// Tiled bf16 GEMM: C[M,N] = A[M,K] @ BT[N,K]^T.
// Block = 256 threads (8 waves), 128x128 output tile, K-chunks of 64.
// TDM path: double-buffered LDS; wave 0 issues 2 tensor_load_to_lds per chunk
// (A tile + B tile) for the NEXT chunk while all 8 waves run 16 WMMAs on the
// current one, then s_wait_tensorcnt + barrier.
// ---------------------------------------------------------------------------
template <bool OUT_BF16>
__global__ __launch_bounds__(256) void gemm_bf16_kernel(
    const bf16_t* __restrict__ A,   // [M][K]
    const bf16_t* __restrict__ BT,  // [N][K]
    void* __restrict__ Cout, int M, int N, int K) {
  __shared__ bf16_t la[2][128][64];    // 2 x 16 KB
  __shared__ bf16_t lbT[2][128][64];   // 2 x 16 KB
  const int t    = threadIdx.x;
  const int lane = t & 31;
  const int wave = t >> 5;          // 0..7
  const int wm   = wave & 3;        // 4 M-stripes of 32 rows
  const int wn   = wave >> 2;       // 2 N-halves of 64 cols
  const int bm   = blockIdx.y * 128;
  const int bn   = blockIdx.x * 128;

  v8f acc[2][4] = {};

  auto compute = [&](int buf) {
    v16bf a00 = load_a16x32(&la[buf][wm * 32][0],       64);
    v16bf a10 = load_a16x32(&la[buf][wm * 32 + 16][0],  64);
    v16bf a01 = load_a16x32(&la[buf][wm * 32][32],      64);
    v16bf a11 = load_a16x32(&la[buf][wm * 32 + 16][32], 64);
#pragma unroll
    for (int j = 0; j < 4; ++j) {
      v16bf b0 = load_b_nmajor(&lbT[buf][wn * 64 + j * 16][0],  64);
      v16bf b1 = load_b_nmajor(&lbT[buf][wn * 64 + j * 16][32], 64);
      acc[0][j] = wmma_bf16(a00, b0, acc[0][j]);
      acc[0][j] = wmma_bf16(a01, b1, acc[0][j]);
      acc[1][j] = wmma_bf16(a10, b0, acc[1][j]);
      acc[1][j] = wmma_bf16(a11, b1, acc[1][j]);
    }
  };

#if HAVE_TDM
  const int nk = K >> 6;
  if (wave == 0) {
    tdm_load_tile(&la[0][0][0],  A  + (size_t)bm * K, K, 128);
    tdm_load_tile(&lbT[0][0][0], BT + (size_t)bn * K, K, 128);
    tdm_wait0();
  }
  __syncthreads();
  int buf = 0;
  for (int ik = 0; ik < nk; ++ik) {
    if (ik + 1 < nk && wave == 0) {   // prefetch next chunk via TDM
      const int kkn = (ik + 1) * 64;
      tdm_load_tile(&la[buf ^ 1][0][0],  A  + (size_t)bm * K + kkn, K, 128);
      tdm_load_tile(&lbT[buf ^ 1][0][0], BT + (size_t)bn * K + kkn, K, 128);
    }
    compute(buf);
    if (wave == 0) tdm_wait0();
    __syncthreads();
    buf ^= 1;
  }
#else
  const int srow = t >> 1;          // staging: 0..127
  const int scol = (t & 1) * 32;    // staging: 0 or 32 (halves)
  for (int kk = 0; kk < K; kk += 64) {
    const bf16_t* ga = A  + (size_t)(bm + srow) * K + kk + scol;
    const bf16_t* gb = BT + (size_t)(bn + srow) * K + kk + scol;
#if HAVE_ASYNC_LDS
    async_cp16(ga,      &la[0][srow][scol]);
    async_cp16(ga + 8,  &la[0][srow][scol + 8]);
    async_cp16(ga + 16, &la[0][srow][scol + 16]);
    async_cp16(ga + 24, &la[0][srow][scol + 24]);
    async_cp16(gb,      &lbT[0][srow][scol]);
    async_cp16(gb + 8,  &lbT[0][srow][scol + 8]);
    async_cp16(gb + 16, &lbT[0][srow][scol + 16]);
    async_cp16(gb + 24, &lbT[0][srow][scol + 24]);
    asm volatile("s_wait_asynccnt 0" ::: "memory");
#else
    *(v8bf*)&la[0][srow][scol]       = *(const v8bf*)(ga);
    *(v8bf*)&la[0][srow][scol + 8]   = *(const v8bf*)(ga + 8);
    *(v8bf*)&la[0][srow][scol + 16]  = *(const v8bf*)(ga + 16);
    *(v8bf*)&la[0][srow][scol + 24]  = *(const v8bf*)(ga + 24);
    *(v8bf*)&lbT[0][srow][scol]      = *(const v8bf*)(gb);
    *(v8bf*)&lbT[0][srow][scol + 8]  = *(const v8bf*)(gb + 8);
    *(v8bf*)&lbT[0][srow][scol + 16] = *(const v8bf*)(gb + 16);
    *(v8bf*)&lbT[0][srow][scol + 24] = *(const v8bf*)(gb + 24);
#endif
    __syncthreads();
    compute(0);
    __syncthreads();
  }
#endif

#pragma unroll
  for (int mi = 0; mi < 2; ++mi) {
    const int mrow = bm + wm * 32 + mi * 16 + ((lane & 16) ? 8 : 0);
#pragma unroll
    for (int j = 0; j < 4; ++j) {
      const int ncol = bn + wn * 64 + j * 16 + (lane & 15);
#pragma unroll
      for (int r = 0; r < 8; ++r) {
        size_t o = (size_t)(mrow + r) * N + ncol;
        if constexpr (OUT_BF16) ((bf16_t*)Cout)[o] = f2bf(acc[mi][j][r]);
        else                    ((float*)Cout)[o]  = acc[mi][j][r];
      }
    }
  }
}

// ---------------------------------------------------------------------------
// Flash attention + fused per-head FFN(SiLU).
// grid = (S/64, H); block = 128 (4 independent waves, one 16-query tile each).
// q: [H][S][64], k: [KVH][S][64], vT: [KVH][64][S], fwT: [64][64], fb: [64]
// aout: [S][H*64] bf16
// ---------------------------------------------------------------------------
__global__ __launch_bounds__(128) void attn_kernel(
    const bf16_t* __restrict__ q, const bf16_t* __restrict__ k,
    const bf16_t* __restrict__ vT, const bf16_t* __restrict__ fwT,
    const float* __restrict__ fb, bf16_t* __restrict__ aout) {
  __shared__ bf16_t lpo[4][16][64];   // per-wave P / O staging
  const int t    = threadIdx.x;
  const int lane = t & 31;
  const int wave = t >> 5;
  const int h    = blockIdx.y;
  const int kvh  = h >> 2;            // N_REP = 4
  const int qt   = blockIdx.x * 4 + wave;
  const int qs   = qt * 16;
  const int nn   = lane & 15;
  const int mofs = (lane & 16) ? 8 : 0;
  const float scale = 0.125f;         // 1/sqrt(64)

  const bf16_t* qp = q + ((size_t)h * S_LEN + qs) * HEADDIM;
  v16bf aq0 = load_a16x32(qp, HEADDIM);
  v16bf aq1 = load_a16x32(qp + 32, HEADDIM);

  float mrow[8], lrow[8];
#pragma unroll
  for (int r = 0; r < 8; ++r) { mrow[r] = -3.0e38f; lrow[r] = 0.f; }
  v8f o0 = {}, o1 = {}, o2 = {}, o3 = {};

  for (int n0 = 0; n0 < qs + 16; n0 += 32) {
    // ---- scores S = Q K^T for 32 keys (two 16x16 tiles) ----
    const bf16_t* kp  = k + ((size_t)kvh * S_LEN + n0) * HEADDIM;
    const bf16_t* kp1 = kp + 16 * HEADDIM;
    v8f s0 = {}, s1 = {};
    s0 = wmma_bf16(aq0, load_b_nmajor(kp, HEADDIM), s0);
    s0 = wmma_bf16(aq1, load_b_nmajor(kp + 32, HEADDIM), s0);
    s1 = wmma_bf16(aq0, load_b_nmajor(kp1, HEADDIM), s1);
    s1 = wmma_bf16(aq1, load_b_nmajor(kp1 + 32, HEADDIM), s1);

    // ---- causal mask + online softmax (row stats per lane-half) ----
    const int key0 = n0 + nn;
    const int key1 = key0 + 16;
    float p0[8], p1[8];
#pragma unroll
    for (int r = 0; r < 8; ++r) {
      const int qrow = qs + mofs + r;
      float v0 = s0[r] * scale + ((key0 <= qrow) ? 0.f : -1.0e9f);
      float v1 = s1[r] * scale + ((key1 <= qrow) ? 0.f : -1.0e9f);
      float rm = fmaxf(v0, v1);
      rm = fmaxf(rm, __shfl_xor(rm, 1));
      rm = fmaxf(rm, __shfl_xor(rm, 2));
      rm = fmaxf(rm, __shfl_xor(rm, 4));
      rm = fmaxf(rm, __shfl_xor(rm, 8));
      const float nm    = fmaxf(mrow[r], rm);
      const float alpha = __expf(mrow[r] - nm);
      float e0 = __expf(v0 - nm);
      float e1 = __expf(v1 - nm);
      float ps = e0 + e1;
      ps += __shfl_xor(ps, 1);
      ps += __shfl_xor(ps, 2);
      ps += __shfl_xor(ps, 4);
      ps += __shfl_xor(ps, 8);
      lrow[r] = lrow[r] * alpha + ps;
      mrow[r] = nm;
      o0[r] *= alpha; o1[r] *= alpha; o2[r] *= alpha; o3[r] *= alpha;
      p0[r] = e0; p1[r] = e1;
    }

    // ---- bounce P through LDS into A-fragment layout ----
#pragma unroll
    for (int r = 0; r < 8; ++r) {
      lpo[wave][mofs + r][nn]      = f2bf(p0[r]);
      lpo[wave][mofs + r][nn + 16] = f2bf(p1[r]);
    }
    asm volatile("s_wait_dscnt 0" ::: "memory");
    v16bf ap = load_a16x32(&lpo[wave][0][0], 64);

    // ---- O += P V (vT is d-major so B-frags are contiguous) ----
    const bf16_t* vp = vT + (size_t)kvh * HEADDIM * S_LEN + n0;
    o0 = wmma_bf16(ap, load_b_nmajor(vp, S_LEN), o0);
    o1 = wmma_bf16(ap, load_b_nmajor(vp + (size_t)16 * S_LEN, S_LEN), o1);
    o2 = wmma_bf16(ap, load_b_nmajor(vp + (size_t)32 * S_LEN, S_LEN), o2);
    o3 = wmma_bf16(ap, load_b_nmajor(vp + (size_t)48 * S_LEN, S_LEN), o3);
  }

#pragma unroll
  for (int r = 0; r < 8; ++r) {
    const float inv = 1.0f / lrow[r];
    o0[r] *= inv; o1[r] *= inv; o2[r] *= inv; o3[r] *= inv;
  }

  // ---- fused per-head FFN: y = silu(O @ fw + fb) ----
#pragma unroll
  for (int r = 0; r < 8; ++r) {
    lpo[wave][mofs + r][nn]      = f2bf(o0[r]);
    lpo[wave][mofs + r][nn + 16] = f2bf(o1[r]);
    lpo[wave][mofs + r][nn + 32] = f2bf(o2[r]);
    lpo[wave][mofs + r][nn + 48] = f2bf(o3[r]);
  }
  asm volatile("s_wait_dscnt 0" ::: "memory");
  v16bf ao0 = load_a16x32(&lpo[wave][0][0], 64);
  v16bf ao1 = load_a16x32(&lpo[wave][0][0] + 32, 64);

  v8f y0, y1, y2, y3;
  const float bb0 = fb[nn], bb1 = fb[nn + 16], bb2 = fb[nn + 32], bb3 = fb[nn + 48];
#pragma unroll
  for (int r = 0; r < 8; ++r) { y0[r] = bb0; y1[r] = bb1; y2[r] = bb2; y3[r] = bb3; }
  y0 = wmma_bf16(ao0, load_b_nmajor(fwT + 0 * 16 * 64, 64), y0);
  y0 = wmma_bf16(ao1, load_b_nmajor(fwT + 0 * 16 * 64 + 32, 64), y0);
  y1 = wmma_bf16(ao0, load_b_nmajor(fwT + 1 * 16 * 64, 64), y1);
  y1 = wmma_bf16(ao1, load_b_nmajor(fwT + 1 * 16 * 64 + 32, 64), y1);
  y2 = wmma_bf16(ao0, load_b_nmajor(fwT + 2 * 16 * 64, 64), y2);
  y2 = wmma_bf16(ao1, load_b_nmajor(fwT + 2 * 16 * 64 + 32, 64), y2);
  y3 = wmma_bf16(ao0, load_b_nmajor(fwT + 3 * 16 * 64, 64), y3);
  y3 = wmma_bf16(ao1, load_b_nmajor(fwT + 3 * 16 * 64 + 32, 64), y3);

  bf16_t* op = aout + (size_t)(qs + mofs) * D_MODEL + h * HEADDIM + nn;
#pragma unroll
  for (int r = 0; r < 8; ++r) {
    float u0 = y0[r], u1 = y1[r], u2 = y2[r], u3 = y3[r];
    op[(size_t)r * D_MODEL + 0]  = f2bf(u0 / (1.f + __expf(-u0)));
    op[(size_t)r * D_MODEL + 16] = f2bf(u1 / (1.f + __expf(-u1)));
    op[(size_t)r * D_MODEL + 32] = f2bf(u2 / (1.f + __expf(-u2)));
    op[(size_t)r * D_MODEL + 48] = f2bf(u3 / (1.f + __expf(-u3)));
  }
}

// ---------------------------------------------------------------------------
// Launcher
// ---------------------------------------------------------------------------
extern "C" void kernel_launch(void* const* d_in, const int* in_sizes, int n_in,
                              void* d_out, int out_size, void* d_ws, size_t ws_size,
                              hipStream_t stream) {
  (void)in_sizes; (void)n_in; (void)out_size; (void)ws_size;
  const float* x    = (const float*)d_in[0];
  const float* fcos = (const float*)d_in[1];
  const float* fsin = (const float*)d_in[2];
  // d_in[3] mask: causal mask is computed analytically in the kernel
  const float* wq = (const float*)d_in[4];
  const float* wk = (const float*)d_in[5];
  const float* wv = (const float*)d_in[6];
  const float* wo = (const float*)d_in[7];
  const float* fw = (const float*)d_in[8];
  const float* fb = (const float*)d_in[9];
  float* out = (float*)d_out;

  const size_t MB = 1024 * 1024;
  char* ws = (char*)d_ws;
  bf16_t* xb    = (bf16_t*)(ws + 0 * MB);   // [S][D]            8 MB
  bf16_t* wqT   = (bf16_t*)(ws + 8 * MB);   // [D][D]   N-major  8 MB
  bf16_t* wkT   = (bf16_t*)(ws + 16 * MB);  // [512][D] N-major  2 MB
  bf16_t* wvT   = (bf16_t*)(ws + 18 * MB);  // [512][D] N-major  2 MB
  bf16_t* woT   = (bf16_t*)(ws + 20 * MB);  // [D][D]   N-major  8 MB
  bf16_t* qtmp  = (bf16_t*)(ws + 28 * MB);  // [S][D]            8 MB
  bf16_t* ktmp  = (bf16_t*)(ws + 36 * MB);  // [S][512]          2 MB
  bf16_t* vtmp  = (bf16_t*)(ws + 38 * MB);  // [S][512]          2 MB
  bf16_t* qrope = (bf16_t*)(ws + 40 * MB);  // [H][S][64]        8 MB
  bf16_t* krope = (bf16_t*)(ws + 48 * MB);  // [KVH][S][64]      2 MB
  bf16_t* vT    = (bf16_t*)(ws + 50 * MB);  // [KVH][64][S]      2 MB
  bf16_t* attn  = (bf16_t*)(ws + 52 * MB);  // [S][H*64]         8 MB
  bf16_t* fwT   = (bf16_t*)(ws + 60 * MB);  // [64][64]          8 KB

  const int nX = S_LEN * D_MODEL;
  f32_to_bf16_kernel<<<(nX + 255) / 256, 256, 0, stream>>>(x, xb, nX);
  // weights -> N-major bf16 (one-time transposed convert, LDS tiled)
  transpose_f32_to_bf16_kernel<<<dim3(D_MODEL / 32, D_MODEL / 32), 256, 0, stream>>>(
      wq, wqT, D_MODEL, D_MODEL);
  transpose_f32_to_bf16_kernel<<<dim3(KVDIM / 32, D_MODEL / 32), 256, 0, stream>>>(
      wk, wkT, D_MODEL, KVDIM);
  transpose_f32_to_bf16_kernel<<<dim3(KVDIM / 32, D_MODEL / 32), 256, 0, stream>>>(
      wv, wvT, D_MODEL, KVDIM);
  transpose_f32_to_bf16_kernel<<<dim3(D_MODEL / 32, D_MODEL / 32), 256, 0, stream>>>(
      wo, woT, D_MODEL, D_MODEL);
  pack_fwT_kernel<<<(HEADDIM * HEADDIM + 255) / 256, 256, 0, stream>>>(fw, fwT);

  // QKV projections (128x128 tiles, TDM double-buffered staging)
  gemm_bf16_kernel<true><<<dim3(D_MODEL / 128, S_LEN / 128), 256, 0, stream>>>(
      xb, wqT, qtmp, S_LEN, D_MODEL, D_MODEL);
  gemm_bf16_kernel<true><<<dim3(KVDIM / 128, S_LEN / 128), 256, 0, stream>>>(
      xb, wkT, ktmp, S_LEN, KVDIM, D_MODEL);
  gemm_bf16_kernel<true><<<dim3(KVDIM / 128, S_LEN / 128), 256, 0, stream>>>(
      xb, wvT, vtmp, S_LEN, KVDIM, D_MODEL);

  // RoPE + repack into head-major layouts
  const int nQr = S_LEN * NHEADS * 32;
  const int nKr = S_LEN * KVHEADS * 32;
  rope_pack_kernel<<<(nQr + 255) / 256, 256, 0, stream>>>(qtmp, fcos, fsin, qrope, NHEADS, nQr);
  rope_pack_kernel<<<(nKr + 255) / 256, 256, 0, stream>>>(ktmp, fcos, fsin, krope, KVHEADS, nKr);
  const int nV = KVHEADS * HEADDIM * S_LEN;
  pack_vT_kernel<<<(nV + 255) / 256, 256, 0, stream>>>(vtmp, vT, nV);

  // flash attention + fused per-head FFN
  attn_kernel<<<dim3(S_LEN / 64, NHEADS), 128, 0, stream>>>(qrope, krope, vT, fwT, fb, attn);

  // output projection (f32 out)
  gemm_bf16_kernel<false><<<dim3(D_MODEL / 128, S_LEN / 128), 256, 0, stream>>>(
      attn, woT, out, S_LEN, D_MODEL, D_MODEL);
}
